// ReLULSTM_16071767621782
// MI455X (gfx1250) — compile-verified, matrix operand order checked by reading, not run on hardware
//
#include <hip/hip_runtime.h>
#include <math.h>

#define BATCH 4096
#define DIM   2048   // INPUT_SIZE == HIDDEN_SIZE == 2048

typedef __attribute__((ext_vector_type(16))) __bf16 bf16x16;
typedef __attribute__((ext_vector_type(8)))  float  f32x8;
typedef __attribute__((ext_vector_type(4)))  unsigned int u32x4;
typedef __attribute__((ext_vector_type(8)))  int    i32x8;
typedef __attribute__((ext_vector_type(4)))  int    i32x4;

union FragU {
    unsigned int u[8];
    bf16x16      v;
};

__device__ __forceinline__ unsigned int f2bf(float f) {
    union { float f; unsigned int u; } x;
    x.f = f;
    // round-to-nearest-even f32 -> bf16
    unsigned int r = x.u + 0x7FFFu + ((x.u >> 16) & 1u);
    return r >> 16;
}

// ---------------------------------------------------------------------------
// Pass 1: one-time f32 -> bf16 conversion (bandwidth bound)
// ---------------------------------------------------------------------------
__global__ __launch_bounds__(256) void cvt_f32_bf16_kernel(
    const float* __restrict__ src, unsigned short* __restrict__ dst, int n4)
{
    int i = blockIdx.x * 256 + threadIdx.x;
    const int stride = gridDim.x * 256;
    for (; i < n4; i += stride) {
        float4 v = reinterpret_cast<const float4*>(src)[i];
        ushort4 o;
        o.x = (unsigned short)f2bf(v.x);
        o.y = (unsigned short)f2bf(v.y);
        o.z = (unsigned short)f2bf(v.z);
        o.w = (unsigned short)f2bf(v.w);
        reinterpret_cast<ushort4*>(dst)[i] = o;
    }
}

// ---------------------------------------------------------------------------
// Pass 2: TDM-fed, double-buffered WMMA GEMM + fused LSTM epilogue
//   block = 256 thr (8 waves, 2x4). Block tile: 64 rows x 64 cols, 4 gates.
//   Waves 0..4 each DMA one tile per K-step via tensor_load_to_lds.
// ---------------------------------------------------------------------------
__global__ __launch_bounds__(256) void relulstm_tdm_wmma_kernel(
    const unsigned short* __restrict__ xb, const unsigned short* __restrict__ hb,
    const unsigned short* __restrict__ wb,   // 8 matrices: [Wfx,Wix,Wcx,Wox, Wfh,Wih,Wch,Woh]
    const float* __restrict__ c,
    const float* __restrict__ bfx, const float* __restrict__ bix,
    const float* __restrict__ bcx, const float* __restrict__ boxb,
    float* __restrict__ outH, float* __restrict__ outC)
{
    // 2 buffers x 5 tiles x (64 rows x 32 K bf16 = 4KB) = 40KB
    __shared__ __align__(16) unsigned short sT[2 * 5 * 64 * 32];

    const int tid   = threadIdx.x;
    const int lane  = tid & 31;
    const int wave  = tid >> 5;
    const int waveM = wave >> 2;    // 0..1
    const int waveN = wave & 3;     // 0..3
    const int ln    = lane & 15;
    const int kh    = lane >> 4;

    const int mBase = blockIdx.y * 64;   // batch rows
    const int nBase = blockIdx.x * 64;   // hidden cols

    const unsigned ldsBase = (unsigned)(size_t)(void*)&sT[0];

    // Issue one TDM 2D tile load for this wave (valid for wave < 5).
    //   wave 0   : A tile   (64 batch rows x 32 K) from xb/hb
    //   wave 1-4 : B tile g (64 W rows    x 32 K) from wb
    auto tdm_issue = [&](int L, int buf) {
        const int p  = L >> 6;           // 0: x-side K, 1: h-side K
        const int k0 = (L & 63) * 32;
        const unsigned short* src;
        unsigned tRows;
        int rowBase;
        if (wave == 0) { src = p ? hb : xb; tRows = BATCH; rowBase = mBase; }
        else {
            src = wb + (size_t)(p * 4 + (wave - 1)) * DIM * DIM;
            tRows = DIM; rowBase = nBase;
        }
        const unsigned long long ga =
            (unsigned long long)(size_t)(const void*)(src + (size_t)rowBase * DIM + k0);
        const unsigned ldsAddr = ldsBase + (unsigned)(buf * 5 + wave) * 4096u;

        // D# group 0 (ISA 08 "async_tensor" sec 8.3)
        u32x4 g0;
        g0[0] = 1u;                                   // count=1, no gather
        g0[1] = ldsAddr;                              // lds_addr
        g0[2] = (unsigned)(ga & 0xFFFFFFFFull);       // global_addr[31:0]
        g0[3] = (unsigned)((ga >> 32) & 0x01FFFFFFull) | (2u << 30);  // ga[56:32] | type=2

        // D# group 1 (sec 8.4)
        i32x8 g1;
        g1[0] = 0x10000;                 // workgroup_mask=0, data_size=1 (2 bytes)
        g1[1] = (int)((unsigned)DIM << 16);   // tensor_dim0[15:0] in [31:16]
        g1[2] = (int)(tRows << 16);           // tensor_dim1[15:0] in [31:16]
        g1[3] = (int)(32u << 16);             // tile_dim0=32 in [31:16]
        g1[4] = 64;                           // tile_dim1=64 (tile_dim2=0)
        g1[5] = (int)DIM;                     // tensor_dim0_stride[31:0]
        g1[6] = 0;                            // stride hi / dim1_stride lo (unused, 2D)
        g1[7] = 0;

        i32x4 z4 = {0, 0, 0, 0};              // groups 2/3 unused (2D tile)
#if defined(__clang_major__) && (__clang_major__ >= 23)
        i32x8 z8 = {0, 0, 0, 0, 0, 0, 0, 0};
        __builtin_amdgcn_tensor_load_to_lds(g0, g1, z4, z4, z8, 0);
#else
        __builtin_amdgcn_tensor_load_to_lds(g0, g1, z4, z4, 0);
#endif
    };

    f32x8 acc[2][4];
    {
        f32x8 zz = {};
        #pragma unroll
        for (int t = 0; t < 2; ++t)
            #pragma unroll
            for (int g = 0; g < 4; ++g)
                acc[t][g] = zz;
    }

    // Prologue: fill buffer 0
    if (wave < 5) tdm_issue(0, 0);

    for (int kb = 0; kb < 2 * DIM / 32; ++kb) {
        const int cbuf = kb & 1;

        __builtin_amdgcn_s_wait_tensorcnt(0);  // this wave's DMA (if any) landed
        __syncthreads();                        // all tiles of cbuf visible; prev reads of other buf done

        // Overlap: kick off next buffer's DMA while we compute this one
        if (wave < 5 && kb + 1 < 2 * DIM / 32) tdm_issue(kb + 1, cbuf ^ 1);

        const uint4* tA = reinterpret_cast<const uint4*>(sT + (size_t)(cbuf * 5) * 2048);
        FragU a0, a1;
        const int ar0 = waveM * 32 + ln;
        const int ar1 = ar0 + 16;
        *reinterpret_cast<uint4*>(&a0.u[0]) = tA[ar0 * 4 + kh];
        *reinterpret_cast<uint4*>(&a0.u[4]) = tA[ar0 * 4 + kh + 2];
        *reinterpret_cast<uint4*>(&a1.u[0]) = tA[ar1 * 4 + kh];
        *reinterpret_cast<uint4*>(&a1.u[4]) = tA[ar1 * 4 + kh + 2];

        const int br = waveN * 16 + ln;
        #pragma unroll
        for (int g = 0; g < 4; ++g) {
            const uint4* tB = reinterpret_cast<const uint4*>(
                sT + (size_t)(cbuf * 5 + 1 + g) * 2048);
            FragU b;
            *reinterpret_cast<uint4*>(&b.u[0]) = tB[br * 4 + kh];
            *reinterpret_cast<uint4*>(&b.u[4]) = tB[br * 4 + kh + 2];
            acc[0][g] = __builtin_amdgcn_wmma_f32_16x16x32_bf16(
                false, a0.v, false, b.v, (short)0, acc[0][g], false, false);
            acc[1][g] = __builtin_amdgcn_wmma_f32_16x16x32_bf16(
                false, a1.v, false, b.v, (short)0, acc[1][g], false, false);
        }
    }

    // Fused epilogue
    const int col = nBase + waveN * 16 + ln;
    const float bF = bfx[col];
    const float bI = bix[col];
    const float bC = bcx[col];
    const float bO = boxb[col];

    #pragma unroll
    for (int t = 0; t < 2; ++t) {
        const int m0 = mBase + waveM * 32 + t * 16 + kh * 8;
        #pragma unroll
        for (int r = 0; r < 8; ++r) {
            const size_t idx = (size_t)(m0 + r) * DIM + col;
            const float fg = fmaxf(acc[t][0][r] + bF, 0.0f);
            const float ig = fmaxf(acc[t][1][r] + bI, 0.0f);
            const float ct = tanhf(acc[t][2][r] + bC);
            const float og = fmaxf(acc[t][3][r] + bO, 0.0f);
            const float cn = fg * c[idx] + ig * ct;
            outH[idx] = og * tanhf(cn);
            outC[idx] = cn;
        }
    }
}

// ---------------------------------------------------------------------------
// Fallback (round-1 style): fused kernel with in-loop conversion, used only
// if the workspace is too small for the bf16 copies.
// ---------------------------------------------------------------------------
__device__ __forceinline__ unsigned long long pack4(const float4 v) {
    unsigned int lo = f2bf(v.x) | (f2bf(v.y) << 16);
    unsigned int hi = f2bf(v.z) | (f2bf(v.w) << 16);
    return (unsigned long long)lo | ((unsigned long long)hi << 32);
}

__global__ __launch_bounds__(256) void relulstm_wmma_fallback_kernel(
    const float* __restrict__ x,   const float* __restrict__ h,
    const float* __restrict__ c,
    const float* __restrict__ Wfx, const float* __restrict__ bfx, const float* __restrict__ Wfh,
    const float* __restrict__ Wix, const float* __restrict__ bix, const float* __restrict__ Wih,
    const float* __restrict__ Wcx, const float* __restrict__ bcx, const float* __restrict__ Wch,
    const float* __restrict__ Wox, const float* __restrict__ boxb, const float* __restrict__ Woh,
    float* __restrict__ outH, float* __restrict__ outC)
{
    __shared__ __align__(16) unsigned long long sA[64 * 8];
    __shared__ __align__(16) unsigned long long sB[4 * 64 * 8];

    const int tid   = threadIdx.x;
    const int lane  = tid & 31;
    const int wave  = tid >> 5;
    const int waveM = wave >> 2;
    const int waveN = wave & 3;
    const int ln    = lane & 15;
    const int kh    = lane >> 4;

    const int mBase = blockIdx.y * 64;
    const int nBase = blockIdx.x * 64;

    f32x8 acc[2][4];
    {
        f32x8 z = {};
        #pragma unroll
        for (int t = 0; t < 2; ++t)
            #pragma unroll
            for (int g = 0; g < 4; ++g)
                acc[t][g] = z;
    }

    const uint4* sA4 = reinterpret_cast<const uint4*>(sA);
    const uint4* sB4 = reinterpret_cast<const uint4*>(sB);

    for (int phase = 0; phase < 2; ++phase) {
        const float* A = phase ? h : x;
        const float* Warr[4];
        Warr[0] = phase ? Wfh : Wfx;
        Warr[1] = phase ? Wih : Wix;
        Warr[2] = phase ? Wch : Wcx;
        Warr[3] = phase ? Woh : Wox;

        for (int kb = 0; kb < DIM / 32; ++kb) {
            const int k0 = kb * 32;
            __syncthreads();
            #pragma unroll
            for (int s = 0; s < 2; ++s) {
                const int slot = tid + s * 256;
                const int row  = slot >> 3;
                const int ch   = slot & 7;
                float4 v = *reinterpret_cast<const float4*>(
                    A + (size_t)(mBase + row) * DIM + k0 + ch * 4);
                sA[row * 8 + ch] = pack4(v);
            }
            #pragma unroll
            for (int g = 0; g < 4; ++g) {
                #pragma unroll
                for (int s = 0; s < 2; ++s) {
                    const int slot = tid + s * 256;
                    const int row  = slot >> 3;
                    const int ch   = slot & 7;
                    float4 v = *reinterpret_cast<const float4*>(
                        Warr[g] + (size_t)(nBase + row) * DIM + k0 + ch * 4);
                    sB[(g * 64 + row) * 8 + ch] = pack4(v);
                }
            }
            __syncthreads();

            FragU a0, a1;
            const int ar0 = waveM * 32 + ln;
            const int ar1 = ar0 + 16;
            *reinterpret_cast<uint4*>(&a0.u[0]) = sA4[ar0 * 4 + kh];
            *reinterpret_cast<uint4*>(&a0.u[4]) = sA4[ar0 * 4 + kh + 2];
            *reinterpret_cast<uint4*>(&a1.u[0]) = sA4[ar1 * 4 + kh];
            *reinterpret_cast<uint4*>(&a1.u[4]) = sA4[ar1 * 4 + kh + 2];

            const int br = waveN * 16 + ln;
            #pragma unroll
            for (int g = 0; g < 4; ++g) {
                FragU b;
                *reinterpret_cast<uint4*>(&b.u[0]) = sB4[(g * 64 + br) * 4 + kh];
                *reinterpret_cast<uint4*>(&b.u[4]) = sB4[(g * 64 + br) * 4 + kh + 2];
                acc[0][g] = __builtin_amdgcn_wmma_f32_16x16x32_bf16(
                    false, a0.v, false, b.v, (short)0, acc[0][g], false, false);
                acc[1][g] = __builtin_amdgcn_wmma_f32_16x16x32_bf16(
                    false, a1.v, false, b.v, (short)0, acc[1][g], false, false);
            }
        }
    }

    const int col = nBase + waveN * 16 + ln;
    const float bF = bfx[col];
    const float bI = bix[col];
    const float bC = bcx[col];
    const float bO = boxb[col];

    #pragma unroll
    for (int t = 0; t < 2; ++t) {
        const int m0 = mBase + waveM * 32 + t * 16 + kh * 8;
        #pragma unroll
        for (int r = 0; r < 8; ++r) {
            const size_t idx = (size_t)(m0 + r) * DIM + col;
            const float fg = fmaxf(acc[t][0][r] + bF, 0.0f);
            const float ig = fmaxf(acc[t][1][r] + bI, 0.0f);
            const float ct = tanhf(acc[t][2][r] + bC);
            const float og = fmaxf(acc[t][3][r] + bO, 0.0f);
            const float cn = fg * c[idx] + ig * ct;
            outH[idx] = og * tanhf(cn);
            outC[idx] = cn;
        }
    }
}

// ---------------------------------------------------------------------------
extern "C" void kernel_launch(void* const* d_in, const int* in_sizes, int n_in,
                              void* d_out, int out_size, void* d_ws, size_t ws_size,
                              hipStream_t stream) {
    (void)in_sizes; (void)n_in; (void)out_size;
    const float* x    = (const float*)d_in[0];
    const float* h    = (const float*)d_in[1];
    const float* c    = (const float*)d_in[2];
    const float* Wfx  = (const float*)d_in[3];
    const float* bfx  = (const float*)d_in[4];
    const float* Wfh  = (const float*)d_in[5];
    const float* Wix  = (const float*)d_in[6];
    const float* bix  = (const float*)d_in[7];
    const float* Wih  = (const float*)d_in[8];
    const float* Wcx  = (const float*)d_in[9];
    const float* bcx  = (const float*)d_in[10];
    const float* Wch  = (const float*)d_in[11];
    const float* Wox  = (const float*)d_in[12];
    const float* boxb = (const float*)d_in[13];
    const float* Woh  = (const float*)d_in[14];

    float* outH = (float*)d_out;
    float* outC = outH + (size_t)BATCH * DIM;

    const size_t xe = (size_t)BATCH * DIM;   // elements in x / h
    const size_t we = (size_t)DIM * DIM;     // elements per weight matrix
    const size_t needed = (2 * xe + 8 * we) * sizeof(unsigned short);  // ~96 MiB

    dim3 grid(DIM / 64, BATCH / 64);         // 32 x 64 = 2048 workgroups

    if (ws_size >= needed) {
        unsigned short* xb = (unsigned short*)d_ws;
        unsigned short* hb = xb + xe;
        unsigned short* wbp = hb + xe;       // 8 matrices, phase-major

        const float* wsrc[8] = {Wfx, Wix, Wcx, Wox, Wfh, Wih, Wch, Woh};
        {
            int n4 = (int)(xe / 4);
            cvt_f32_bf16_kernel<<<4096, 256, 0, stream>>>(x, xb, n4);
            cvt_f32_bf16_kernel<<<4096, 256, 0, stream>>>(h, hb, n4);
        }
        for (int i = 0; i < 8; ++i) {
            int n4 = (int)(we / 4);
            cvt_f32_bf16_kernel<<<4096, 256, 0, stream>>>(wsrc[i], wbp + (size_t)i * we, n4);
        }

        relulstm_tdm_wmma_kernel<<<grid, 256, 0, stream>>>(
            xb, hb, wbp, c, bfx, bix, bcx, boxb, outH, outC);
    } else {
        relulstm_wmma_fallback_kernel<<<grid, 256, 0, stream>>>(
            x, h, c, Wfx, bfx, Wfh, Wix, bix, Wih, Wcx, bcx, Wch, Wox, boxb, Woh,
            outH, outC);
    }
}
